// Gru_cond_layer_27719718929052
// MI455X (gfx1250) — matrix operand order repaired and checked by reading, not compiled, not run on hardware
//
#include <hip/hip_runtime.h>
#include <hip/hip_bf16.h>
#include <math.h>

// ---- dims ----
#define T_  8
#define B_  8
#define M_  256
#define N_  256
#define D_  684
#define A_  512
#define H_  16
#define W_  40
#define QC_ 512
#define K_  11
#define KK_ 121
#define HW_ (H_*W_)          // 640
#define HWB_ (HW_*B_)        // 5120
#define HP_ (H_+2*5)         // 26
#define WP_ (W_+2*5)         // 50
#define APPAD_ (B_*HP_*WP_)  // 10400

#define MT_   (HWB_/16)      // 320 M-tiles
#define NT_   (A_/16)        // 32  N-tiles
#define NKC_UA_ 22           // ceil(684/32)
#define NKC_P_  8            // 256/32

typedef __attribute__((ext_vector_type(16))) _Float16 v16h;
typedef __attribute__((ext_vector_type(8)))  float    v8f;

__device__ __forceinline__ float wred_sum(float v) {
#pragma unroll
  for (int o = 16; o > 0; o >>= 1) v += __shfl_xor(v, o, 32);
  return v;
}
__device__ __forceinline__ float wred_max(float v) {
#pragma unroll
  for (int o = 16; o > 0; o >>= 1) v = fmaxf(v, __shfl_xor(v, o, 32));
  return v;
}
__device__ __forceinline__ float sigm(float x) { return 1.f / (1.f + __expf(-x)); }

// ---------------- init: zero padded ap planes, copy init_state -> h ----------------
__global__ void k_init(float* appad_off, float* appad_on, float* h,
                       const float* init_state) {
  int i = blockIdx.x * blockDim.x + threadIdx.x;
  if (i < APPAD_) { appad_off[i] = 0.f; appad_on[i] = 0.f; }
  if (i < B_ * N_) h[i] = init_state[i];
}

// ---- WMMA fragment layout helpers (wave32, 16x16x32 f16) --------------------------
// A element e of lane L: m = L&15, k = 16*(L>>4) + e
// B element e of lane L: n = L&15, k = 16*(L>>4) + e
// C vgpr r of lane L:    m = r + 8*(L>>4), n = L&15

// ---------------- pack context into A-swizzled f16 (K padded 684->704) -------------
// layout: aswz[ ((tm*NKC_UA_ + kc)*32 + lane)*16 + e ]
__global__ void k_pack_a_ctx(const float* __restrict__ ctx, _Float16* __restrict__ aswz) {
  int gid = blockIdx.x * blockDim.x + threadIdx.x;
  if (gid >= MT_ * NKC_UA_ * 32 * 16) return;
  int e    = gid & 15;
  int lane = (gid >> 4) & 31;
  int kc   = (gid >> 9) % NKC_UA_;
  int tm   = gid / (NKC_UA_ * 512);
  int m = tm * 16 + (lane & 15);
  int k = kc * 32 + ((lane >> 4) << 4) + e;
  int b = m & 7, rw = m >> 3;
  float v = (k < D_) ? ctx[(long)b * D_ * HW_ + (long)k * HW_ + rw] : 0.f;
  aswz[gid] = (_Float16)v;
}

// ---------------- pack W (K x 512) into B-swizzled f16 -----------------------------
// layout: bswz[ ((tn*nkc + kc)*32 + lane)*16 + e ]
__global__ void k_pack_b(const float* __restrict__ w, _Float16* __restrict__ bswz,
                         int nkc, int ktot) {
  int gid = blockIdx.x * blockDim.x + threadIdx.x;
  if (gid >= NT_ * nkc * 32 * 16) return;
  int e    = gid & 15;
  int lane = (gid >> 4) & 31;
  int kc   = (gid >> 9) % nkc;
  int tn   = gid / (nkc * 512);
  int k = kc * 32 + ((lane >> 4) << 4) + e;
  int n = tn * 16 + (lane & 15);
  float v = (k < ktot) ? w[k * A_ + n] : 0.f;
  bswz[gid] = (_Float16)v;
}

// ---------------- merged coverage kernel, written B-swizzled -----------------------
// logical Wm[k][a], k in [0,256): k<128 -> off tap t=k (valid t<121), k>=128 -> on
__global__ void k_wm(const float* __restrict__ wq_off, const float* __restrict__ wq_on,
                     const float* __restrict__ wuf_off, const float* __restrict__ wuf_on,
                     _Float16* __restrict__ wm_swz) {
  int gid = blockIdx.x * blockDim.x + threadIdx.x;
  if (gid >= 256 * A_) return;
  int k = gid >> 9, a = gid & (A_ - 1);
  int t = k & 127;
  float v = 0.f;
  if (t < KK_) {
    const float* wq  = (k < 128) ? wq_off  : wq_on;
    const float* wuf = (k < 128) ? wuf_off : wuf_on;
    for (int c = 0; c < QC_; ++c) v += wq[c * KK_ + t] * wuf[c * A_ + a];
  }
  int tn = a >> 4, kc = k >> 5, hi = (k >> 4) & 1, e = k & 15;
  int lane = hi * 16 + (a & 15);
  wm_swz[((tn * NKC_P_ + kc) * 32 + lane) * 16 + e] = (_Float16)v;
}

// ---------------- per-step: pack dual coverage patches A-swizzled ------------------
// logical P[row][k], row=hw*8+b, k<128: off tap k, k>=128: on tap k-128 (tap<121)
__global__ void k_pack_patch(const float* __restrict__ appad_off,
                             const float* __restrict__ appad_on,
                             _Float16* __restrict__ pswz) {
  int gid = blockIdx.x * blockDim.x + threadIdx.x;
  if (gid >= MT_ * NKC_P_ * 32 * 16) return;
  int e    = gid & 15;
  int lane = (gid >> 4) & 31;
  int kc   = (gid >> 9) % NKC_P_;
  int tm   = gid / (NKC_P_ * 512);
  int m = tm * 16 + (lane & 15);
  int k = kc * 32 + ((lane >> 4) << 4) + e;
  int b = m & 7, rw = m >> 3;
  int hh = rw / W_, ww = rw % W_;
  int t = k & 127;
  float v = 0.f;
  if (t < KK_) {
    const float* ap = (k < 128) ? appad_off : appad_on;
    v = ap[b * (HP_ * WP_) + (hh + t / K_) * WP_ + (ww + t % K_)];
  }
  pswz[gid] = (_Float16)v;
}

// ---------------- Ua GEMM: ua[m*512+n] = sum_k A[m,k]B[k,n] + bias[n] --------------
// one wave: 1 M-tile x 4 N-tiles; clean swizzled v16h loads.
__global__ void k_ua_gemm(const _Float16* __restrict__ aswz,
                          const _Float16* __restrict__ bswz,
                          const float* __restrict__ bua, float* __restrict__ ua) {
  int wave = (blockIdx.x * blockDim.x + threadIdx.x) >> 5;
  if (wave >= MT_ * (NT_ / 4)) return;
  int lane = threadIdx.x & 31;
  int tm = wave >> 3;
  int tg = wave & 7;                     // group of 4 N-tiles
  int hi = lane >> 4;
  const v16h* ap = (const v16h*)aswz + (long)tm * NKC_UA_ * 32 + lane;
  const v16h* b0 = (const v16h*)bswz + (long)(tg * 4 + 0) * NKC_UA_ * 32 + lane;
  const v16h* b1 = (const v16h*)bswz + (long)(tg * 4 + 1) * NKC_UA_ * 32 + lane;
  const v16h* b2 = (const v16h*)bswz + (long)(tg * 4 + 2) * NKC_UA_ * 32 + lane;
  const v16h* b3 = (const v16h*)bswz + (long)(tg * 4 + 3) * NKC_UA_ * 32 + lane;
  v8f c0 = {}, c1 = {}, c2 = {}, c3 = {};
  for (int kc = 0; kc < NKC_UA_; ++kc) {
    v16h av = ap[kc * 32];
    if (kc + 1 < NKC_UA_) __builtin_prefetch(ap + (kc + 1) * 32, 0, 0);
    c0 = __builtin_amdgcn_wmma_f32_16x16x32_f16(false, av, false, b0[kc * 32],
                                                (short)0, c0, false, false);
    c1 = __builtin_amdgcn_wmma_f32_16x16x32_f16(false, av, false, b1[kc * 32],
                                                (short)0, c1, false, false);
    c2 = __builtin_amdgcn_wmma_f32_16x16x32_f16(false, av, false, b2[kc * 32],
                                                (short)0, c2, false, false);
    c3 = __builtin_amdgcn_wmma_f32_16x16x32_f16(false, av, false, b3[kc * 32],
                                                (short)0, c3, false, false);
  }
#pragma unroll
  for (int i = 0; i < 4; ++i) {
    v8f acc = (i == 0) ? c0 : (i == 1) ? c1 : (i == 2) ? c2 : c3;
    int ncol = (tg * 4 + i) * 16 + (lane & 15);
#pragma unroll
    for (int r = 0; r < 8; ++r) {
      int m = tm * 16 + r + (hi << 3);
      ua[(long)m * A_ + ncol] = acc[r] + bua[ncol];
    }
  }
}

// ---------------- base GEMM (merged dual coverage conv) via WMMA -------------------
// base[m*512+n] = sum_{k<256} P[m,k]*Wm[k,n] + wah1[b,n] + buf_off[n] + buf_on[n]
__global__ void k_base_gemm(const _Float16* __restrict__ pswz,
                            const _Float16* __restrict__ wm_swz,
                            const float* __restrict__ wah1,
                            const float* __restrict__ buf_off,
                            const float* __restrict__ buf_on,
                            float* __restrict__ base) {
  int wave = (blockIdx.x * blockDim.x + threadIdx.x) >> 5;
  if (wave >= MT_ * (NT_ / 4)) return;
  int lane = threadIdx.x & 31;
  int tm = wave >> 3;
  int tg = wave & 7;
  int hi = lane >> 4;
  const v16h* ap = (const v16h*)pswz + (long)tm * NKC_P_ * 32 + lane;
  const v16h* b0 = (const v16h*)wm_swz + (long)(tg * 4 + 0) * NKC_P_ * 32 + lane;
  const v16h* b1 = (const v16h*)wm_swz + (long)(tg * 4 + 1) * NKC_P_ * 32 + lane;
  const v16h* b2 = (const v16h*)wm_swz + (long)(tg * 4 + 2) * NKC_P_ * 32 + lane;
  const v16h* b3 = (const v16h*)wm_swz + (long)(tg * 4 + 3) * NKC_P_ * 32 + lane;
  v8f c0 = {}, c1 = {}, c2 = {}, c3 = {};
#pragma unroll
  for (int kc = 0; kc < NKC_P_; ++kc) {
    v16h av = ap[kc * 32];
    c0 = __builtin_amdgcn_wmma_f32_16x16x32_f16(false, av, false, b0[kc * 32],
                                                (short)0, c0, false, false);
    c1 = __builtin_amdgcn_wmma_f32_16x16x32_f16(false, av, false, b1[kc * 32],
                                                (short)0, c1, false, false);
    c2 = __builtin_amdgcn_wmma_f32_16x16x32_f16(false, av, false, b2[kc * 32],
                                                (short)0, c2, false, false);
    c3 = __builtin_amdgcn_wmma_f32_16x16x32_f16(false, av, false, b3[kc * 32],
                                                (short)0, c3, false, false);
  }
#pragma unroll
  for (int i = 0; i < 4; ++i) {
    v8f acc = (i == 0) ? c0 : (i == 1) ? c1 : (i == 2) ? c2 : c3;
    int ncol = (tg * 4 + i) * 16 + (lane & 15);
#pragma unroll
    for (int r = 0; r < 8; ++r) {
      int m = tm * 16 + r + (hi << 3);
      int bb = m & 7;
      base[(long)m * A_ + ncol] =
          acc[r] + wah1[bb * A_ + ncol] + buf_off[ncol] + buf_on[ncol];
    }
  }
}

// ---------------- input projections sz/sr/sh [T,B,N] ----------------
__global__ void k_szrsh(const float* __restrict__ emb,
                        const float* __restrict__ wz, const float* __restrict__ bz,
                        const float* __restrict__ wr, const float* __restrict__ br,
                        const float* __restrict__ wh, const float* __restrict__ bh,
                        float* __restrict__ sz, float* __restrict__ sr,
                        float* __restrict__ sh) {
  int wave = (blockIdx.x * blockDim.x + threadIdx.x) >> 5;
  if (wave >= T_ * B_ * N_) return;
  int lane = threadIdx.x & 31;
  int n = wave & (N_ - 1);
  int tb = wave >> 8;
  const float* e = emb + tb * M_;
  float az = 0.f, ar = 0.f, ah = 0.f;
  for (int k = lane; k < M_; k += 32) {
    float ev = e[k];
    az += ev * wz[k * N_ + n];
    ar += ev * wr[k * N_ + n];
    ah += ev * wh[k * N_ + n];
  }
  az = wred_sum(az); ar = wred_sum(ar); ah = wred_sum(ah);
  if (lane == 0) {
    sz[wave] = az + bz[n]; sr[wave] = ar + br[n]; sh[wave] = ah + bh[n];
  }
}

// ---------------- GRU1 ----------------
__global__ void k_gru1(const float* __restrict__ h, const float* __restrict__ uhz,
                       const float* __restrict__ uhr, const float* __restrict__ uhh,
                       const float* __restrict__ sz, const float* __restrict__ sr,
                       const float* __restrict__ sh, const float* __restrict__ mask,
                       int t, float* __restrict__ h1) {
  int wave = (blockIdx.x * blockDim.x + threadIdx.x) >> 5;
  if (wave >= B_ * N_) return;
  int lane = threadIdx.x & 31;
  int n = wave & (N_ - 1), b = wave >> 8;
  float az = 0.f, ar = 0.f, ah = 0.f;
  const float* hb = h + b * N_;
  for (int k = lane; k < N_; k += 32) {
    float hv = hb[k];
    az += hv * uhz[k * N_ + n];
    ar += hv * uhr[k * N_ + n];
    ah += hv * uhh[k * N_ + n];
  }
  az = wred_sum(az); ar = wred_sum(ar); ah = wred_sum(ah);
  if (lane == 0) {
    int idx = t * B_ * N_ + b * N_ + n;
    float z1 = sigm(az + sz[idx]);
    float r1 = sigm(ar + sr[idx]);
    float hp = tanhf(ah * r1 + sh[idx]);
    float hv = hb[n];
    float hn = z1 * hv + (1.f - z1) * hp;
    float m = mask[t * B_ + b];
    h1[b * N_ + n] = m * hn + (1.f - m) * hv;
  }
}

// ---------------- Wa_h1 = h1 @ W_Wa_off  [B, A] ----------------
__global__ void k_wah1(const float* __restrict__ h1, const float* __restrict__ wwa,
                       float* __restrict__ wah1) {
  int wave = (blockIdx.x * blockDim.x + threadIdx.x) >> 5;
  if (wave >= B_ * A_) return;
  int lane = threadIdx.x & 31;
  int a = wave & (A_ - 1), b = wave >> 9;
  float s = 0.f;
  for (int k = lane; k < N_; k += 32) s += h1[b * N_ + k] * wwa[k * A_ + a];
  s = wred_sum(s);
  if (lane == 0) wah1[b * A_ + a] = s;
}

// ---------------- score: e_x[row] = sum_a tanh(ua_x+base)*va_x[a] + bva_x ----------
__global__ void k_score(const float* __restrict__ base,
                        const float* __restrict__ ua_off, const float* __restrict__ ua_on,
                        const float* __restrict__ va_off, const float* __restrict__ bva_off,
                        const float* __restrict__ va_on,  const float* __restrict__ bva_on,
                        float* __restrict__ e_off, float* __restrict__ e_on) {
  int wave = (blockIdx.x * blockDim.x + threadIdx.x) >> 5;
  if (wave >= HWB_) return;
  int lane = threadIdx.x & 31;
  long ro = (long)wave * A_;
  float eo = 0.f, en = 0.f;
  for (int a = lane; a < A_; a += 32) {
    float bs = base[ro + a];
    eo += tanhf(ua_off[ro + a] + bs) * va_off[a];
    en += tanhf(ua_on[ro + a] + bs) * va_on[a];
  }
  eo = wred_sum(eo); en = wred_sum(en);
  if (lane == 0) { e_off[wave] = eo + bva_off[0]; e_on[wave] = en + bva_on[0]; }
}

// ---------------- softmax + ap update + record alpha/ap outputs -------------------
__global__ void k_softmax(const float* __restrict__ e_off, const float* __restrict__ e_on,
                          const float* __restrict__ cmask,
                          float* __restrict__ a_off, float* __restrict__ a_on,
                          float* __restrict__ appad_off, float* __restrict__ appad_on,
                          float* __restrict__ out_alpha_off, float* __restrict__ out_alpha_on,
                          float* __restrict__ out_ap_off, float* __restrict__ out_ap_on,
                          int t) {
  __shared__ float svals[HWB_];
  __shared__ float lmax[32];
  __shared__ float gmax;
  __shared__ float bsum[B_];
  int br = blockIdx.x;
  const float* e = br ? e_on : e_off;
  float* abuf    = br ? a_on : a_off;
  float* appad   = br ? appad_on : appad_off;
  float* oalpha  = (br ? out_alpha_on : out_alpha_off) + t * B_ * HW_;
  float* oap     = (br ? out_ap_on : out_ap_off) + t * B_ * HW_;
  int tid = threadIdx.x;
  if (tid < B_) bsum[tid] = 0.f;
  float lm = -INFINITY;
  for (int i = tid; i < HWB_; i += 1024) { float v = e[i]; svals[i] = v; lm = fmaxf(lm, v); }
  lm = wred_max(lm);
  if ((tid & 31) == 0) lmax[tid >> 5] = lm;
  __syncthreads();
  if (tid < 32) {
    float v = lmax[tid];
    v = wred_max(v);
    if (tid == 0) gmax = v;
  }
  __syncthreads();
  float gm = gmax;
  for (int i = tid; i < HWB_; i += 1024) {
    int b = i & 7, hw = i >> 3;
    svals[i] = __expf(svals[i] - gm) * cmask[b * HW_ + hw];
  }
  __syncthreads();
  {
    int b = tid >> 7, l = tid & 127;
    float s = 0.f;
    for (int hw = l; hw < HW_; hw += 128) s += svals[hw * B_ + b];
    s = wred_sum(s);
    if ((tid & 31) == 0) atomicAdd(&bsum[b], s);
  }
  __syncthreads();
  for (int i = tid; i < HWB_; i += 1024) {
    int b = i & 7, hw = i >> 3;
    int h = hw / W_, w = hw % W_;
    float av = svals[i] / (bsum[b] + 1e-10f);
    abuf[i] = av;
    int pidx = b * (HP_ * WP_) + (h + 5) * WP_ + (w + 5);
    float nap = appad[pidx] + av;
    appad[pidx] = nap;
    oalpha[b * HW_ + hw] = av;
    oap[b * HW_ + hw] = nap;
  }
}

// ---------------- ct_x[b,d] = sum_hw ctx[b,d,hw] * a[hw*8+b]  (both branches) -----
__global__ void k_ct(const float* __restrict__ ctx_off, const float* __restrict__ ctx_on,
                     const float* __restrict__ a_off, const float* __restrict__ a_on,
                     float* __restrict__ ct_cat) {
  int wave = (blockIdx.x * blockDim.x + threadIdx.x) >> 5;
  if (wave >= 2 * B_ * D_) return;
  int lane = threadIdx.x & 31;
  int br = wave / (B_ * D_);
  int rem = wave % (B_ * D_);
  int b = rem / D_, d = rem % D_;
  const float* ctx = br ? ctx_on : ctx_off;
  const float* ab  = br ? a_on : a_off;
  long cb = (long)b * D_ * HW_ + (long)d * HW_;
  float s = 0.f;
  for (int hw = lane; hw < HW_; hw += 32) s += ctx[cb + hw] * ab[hw * B_ + b];
  s = wred_sum(s);
  if (lane == 0) ct_cat[b * (2 * D_) + br * D_ + d] = s;
}

// ---------------- fuse: ct[b,dd] = cat[b,:] @ W_fuse + b_fuse ----------------
__global__ void k_fuse(const float* __restrict__ ct_cat, const float* __restrict__ wf,
                       const float* __restrict__ bf, float* __restrict__ ct,
                       float* __restrict__ out_ct, int t) {
  int wave = (blockIdx.x * blockDim.x + threadIdx.x) >> 5;
  if (wave >= B_ * D_) return;
  int lane = threadIdx.x & 31;
  int b = wave / D_, dd = wave % D_;
  const float* cat = ct_cat + b * (2 * D_);
  float s = 0.f;
  for (int j = lane; j < 2 * D_; j += 32) s += cat[j] * wf[j * D_ + dd];
  s = wred_sum(s);
  if (lane == 0) {
    float v = s + bf[dd];
    ct[b * D_ + dd] = v;
    out_ct[t * B_ * D_ + b * D_ + dd] = v;
  }
}

// ---------------- GRU2 ----------------
__global__ void k_gru2(const float* __restrict__ ct, const float* __restrict__ h1,
                       const float* __restrict__ wcz, const float* __restrict__ wcr,
                       const float* __restrict__ wch,
                       const float* __restrict__ uhz2, const float* __restrict__ buhz2,
                       const float* __restrict__ uhr2, const float* __restrict__ buhr2,
                       const float* __restrict__ uhh2, const float* __restrict__ buhh2,
                       const float* __restrict__ mask, int t,
                       float* __restrict__ h, float* __restrict__ out_h2) {
  int wave = (blockIdx.x * blockDim.x + threadIdx.x) >> 5;
  if (wave >= B_ * N_) return;
  int lane = threadIdx.x & 31;
  int n = wave & (N_ - 1), b = wave >> 8;
  float c1 = 0.f, c2 = 0.f, c3 = 0.f;
  for (int d = lane; d < D_; d += 32) {
    float cv = ct[b * D_ + d];
    c1 += cv * wcz[d * N_ + n];
    c2 += cv * wcr[d * N_ + n];
    c3 += cv * wch[d * N_ + n];
  }
  float u1 = 0.f, u2 = 0.f, u3 = 0.f;
  for (int k = lane; k < N_; k += 32) {
    float hv = h1[b * N_ + k];
    u1 += hv * uhz2[k * N_ + n];
    u2 += hv * uhr2[k * N_ + n];
    u3 += hv * uhh2[k * N_ + n];
  }
  c1 = wred_sum(c1); c2 = wred_sum(c2); c3 = wred_sum(c3);
  u1 = wred_sum(u1); u2 = wred_sum(u2); u3 = wred_sum(u3);
  if (lane == 0) {
    float z2 = sigm(c1 + u1 + buhz2[n]);
    float r2 = sigm(c2 + u2 + buhr2[n]);
    float hp = tanhf(c3 + (u3 + buhh2[n]) * r2);
    float h1v = h1[b * N_ + n];
    float h2 = z2 * h1v + (1.f - z2) * hp;
    float m = mask[t * B_ + b];
    h2 = m * h2 + (1.f - m) * h1v;
    h[b * N_ + n] = h2;
    out_h2[t * B_ * N_ + b * N_ + n] = h2;
  }
}

extern "C" void kernel_launch(void* const* d_in, const int* in_sizes, int n_in,
                              void* d_out, int out_size, void* d_ws, size_t ws_size,
                              hipStream_t stream) {
  const float* embedding   = (const float*)d_in[0];
  const float* mask        = (const float*)d_in[1];
  const float* ctx_off     = (const float*)d_in[2];
  const float* ctx_on      = (const float*)d_in[3];
  const float* cmask       = (const float*)d_in[4];
  const float* init_state  = (const float*)d_in[5];
  const float* W_Ua_off    = (const float*)d_in[6];
  const float* b_Ua_off    = (const float*)d_in[7];
  const float* W_Ua_on     = (const float*)d_in[8];
  const float* b_Ua_on     = (const float*)d_in[9];
  const float* W_Wa_off    = (const float*)d_in[10];
  const float* W_Qoff      = (const float*)d_in[11];
  const float* W_Qon       = (const float*)d_in[12];
  const float* W_Uf_off    = (const float*)d_in[13];
  const float* b_Uf_off    = (const float*)d_in[14];
  const float* W_Uf_on     = (const float*)d_in[15];
  const float* b_Uf_on     = (const float*)d_in[16];
  const float* W_va_off    = (const float*)d_in[17];
  const float* b_va_off    = (const float*)d_in[18];
  const float* W_va_on     = (const float*)d_in[19];
  const float* b_va_on     = (const float*)d_in[20];
  const float* W_fuse      = (const float*)d_in[21];
  const float* b_fuse      = (const float*)d_in[22];
  const float* W_Wyz       = (const float*)d_in[23];
  const float* b_Wyz       = (const float*)d_in[24];
  const float* W_Wyr       = (const float*)d_in[25];
  const float* b_Wyr       = (const float*)d_in[26];
  const float* W_Wyh       = (const float*)d_in[27];
  const float* b_Wyh       = (const float*)d_in[28];
  const float* W_Uhz       = (const float*)d_in[29];
  const float* W_Uhr       = (const float*)d_in[30];
  const float* W_Uhh       = (const float*)d_in[31];
  const float* W_Wcz       = (const float*)d_in[32];
  const float* W_Wcr       = (const float*)d_in[33];
  const float* W_Wch       = (const float*)d_in[34];
  const float* W_Uhz2      = (const float*)d_in[35];
  const float* b_Uhz2      = (const float*)d_in[36];
  const float* W_Uhr2      = (const float*)d_in[37];
  const float* b_Uhr2      = (const float*)d_in[38];
  const float* W_Uhh2      = (const float*)d_in[39];
  const float* b_Uhh2      = (const float*)d_in[40];

  float* ws = (float*)d_ws;
  const long SZ_UA = (long)HWB_ * A_;     // 2,621,440 floats
  long o = 0;
  float* ua_off    = ws + o;              o += SZ_UA;
  float* ua_on     = ws + o;              o += SZ_UA;
  float* base      = ws + o;              o += SZ_UA;
  float* sz        = ws + o;              o += T_ * B_ * N_;
  float* sr        = ws + o;              o += T_ * B_ * N_;
  float* sh        = ws + o;              o += T_ * B_ * N_;
  float* h         = ws + o;              o += B_ * N_;
  float* h1        = ws + o;              o += B_ * N_;
  float* wah1      = ws + o;              o += B_ * A_;
  float* appad_off = ws + o;              o += APPAD_;
  float* appad_on  = ws + o;              o += APPAD_;
  float* e_off     = ws + o;              o += HWB_;
  float* e_on      = ws + o;              o += HWB_;
  float* a_off     = ws + o;              o += HWB_;
  float* a_on      = ws + o;              o += HWB_;
  float* ct_cat    = ws + o;              o += B_ * 2 * D_;
  float* ct        = ws + o;              o += B_ * D_;
  // f16 swizzled operand buffers (all 32B-aligned: offsets are multiples of 8 floats)
  const long HSZ_A  = (long)MT_ * NKC_UA_ * 32 * 16;  // 3,604,480 halfs
  const long HSZ_B  = (long)NT_ * NKC_UA_ * 32 * 16;  //   360,448 halfs
  const long HSZ_WM = (long)NT_ * NKC_P_ * 32 * 16;   //   131,072 halfs
  const long HSZ_P  = (long)MT_ * NKC_P_ * 32 * 16;   // 1,310,720 halfs
  _Float16* aswz_off = (_Float16*)(ws + o);  o += HSZ_A / 2;
  _Float16* aswz_on  = (_Float16*)(ws + o);  o += HSZ_A / 2;
  _Float16* bswz_off = (_Float16*)(ws + o);  o += HSZ_B / 2;
  _Float16* bswz_on  = (_Float16*)(ws + o);  o += HSZ_B / 2;
  _Float16* wm_swz   = (_Float16*)(ws + o);  o += HSZ_WM / 2;
  _Float16* pswz     = (_Float16*)(ws + o);  o += HSZ_P / 2;

  float* out = (float*)d_out;
  float* out_h2    = out;                          // T*B*N
  float* out_ct    = out_h2 + T_ * B_ * N_;        // T*B*D
  float* out_aoff  = out_ct + T_ * B_ * D_;        // T*B*H*W
  float* out_aon   = out_aoff + T_ * B_ * HW_;
  float* out_apoff = out_aon + T_ * B_ * HW_;
  float* out_apon  = out_apoff + T_ * B_ * HW_;

  // ---- precompute ----
  k_init<<<(APPAD_ + 255) / 256, 256, 0, stream>>>(appad_off, appad_on, h, init_state);
  k_wm<<<(256 * A_) / 256, 256, 0, stream>>>(W_Qoff, W_Qon, W_Uf_off, W_Uf_on, wm_swz);
  k_pack_a_ctx<<<(int)(HSZ_A / 256), 256, 0, stream>>>(ctx_off, aswz_off);
  k_pack_a_ctx<<<(int)(HSZ_A / 256), 256, 0, stream>>>(ctx_on,  aswz_on);
  k_pack_b<<<(int)(HSZ_B / 256), 256, 0, stream>>>(W_Ua_off, bswz_off, NKC_UA_, D_);
  k_pack_b<<<(int)(HSZ_B / 256), 256, 0, stream>>>(W_Ua_on,  bswz_on,  NKC_UA_, D_);
  {
    int waves = MT_ * (NT_ / 4);          // 2560
    k_ua_gemm<<<waves / 8, 256, 0, stream>>>(aswz_off, bswz_off, b_Ua_off, ua_off);
    k_ua_gemm<<<waves / 8, 256, 0, stream>>>(aswz_on,  bswz_on,  b_Ua_on,  ua_on);
  }
  k_szrsh<<<(T_ * B_ * N_) / 8, 256, 0, stream>>>(embedding, W_Wyz, b_Wyz, W_Wyr, b_Wyr,
                                                  W_Wyh, b_Wyh, sz, sr, sh);

  // ---- recurrent steps ----
  for (int t = 0; t < T_; ++t) {
    k_gru1<<<(B_ * N_) / 8, 256, 0, stream>>>(h, W_Uhz, W_Uhr, W_Uhh, sz, sr, sh,
                                              mask, t, h1);
    k_wah1<<<(B_ * A_) / 8, 256, 0, stream>>>(h1, W_Wa_off, wah1);
    k_pack_patch<<<(int)(HSZ_P / 256), 256, 0, stream>>>(appad_off, appad_on, pswz);
    k_base_gemm<<<(MT_ * (NT_ / 4)) / 8, 256, 0, stream>>>(
        pswz, wm_swz, wah1, b_Uf_off, b_Uf_on, base);
    k_score<<<HWB_ / 8, 256, 0, stream>>>(base, ua_off, ua_on, W_va_off, b_va_off,
                                          W_va_on, b_va_on, e_off, e_on);
    k_softmax<<<2, 1024, 0, stream>>>(e_off, e_on, cmask, a_off, a_on,
                                      appad_off, appad_on, out_aoff, out_aon,
                                      out_apoff, out_apon, t);
    k_ct<<<(2 * B_ * D_ + 7) / 8, 256, 0, stream>>>(ctx_off, ctx_on, a_off, a_on, ct_cat);
    k_fuse<<<(B_ * D_ + 7) / 8, 256, 0, stream>>>(ct_cat, W_fuse, b_fuse, ct, out_ct, t);
    k_gru2<<<(B_ * N_) / 8, 256, 0, stream>>>(ct, h1, W_Wcz, W_Wcr, W_Wch,
                                              W_Uhz2, b_Uhz2, W_Uhr2, b_Uhr2,
                                              W_Uhh2, b_Uhh2, mask, t, h, out_h2);
  }
}
